// CrossTransformer_62096637166371
// MI455X (gfx1250) — matrix operand order repaired and checked by reading, not compiled, not run on hardware
//
#include <hip/hip_runtime.h>

#define EPSF 1e-5f

typedef __bf16 bf16_t;
typedef __attribute__((ext_vector_type(16))) __bf16 bf16x16;
typedef __attribute__((ext_vector_type(8)))  float  floatx8;
typedef __attribute__((ext_vector_type(4)))  unsigned u32x4;
typedef __attribute__((ext_vector_type(8)))  int      i32x8;
typedef __attribute__((ext_vector_type(4)))  int      i32x4;

union FragU { bf16x16 v; unsigned u[8]; };
union H2 { bf16_t h[2]; unsigned u; };

static __device__ __forceinline__ unsigned pack2(float a, float b) {
  H2 t; t.h[0] = (bf16_t)a; t.h[1] = (bf16_t)b; return t.u;
}

static __device__ __forceinline__ floatx8 wmma_bf16(bf16x16 a, bf16x16 b, floatx8 c) {
  return __builtin_amdgcn_wmma_f32_16x16x32_bf16(false, a, false, b, (short)0, c, false, false);
}

// A fragment: 16(M) x 32(K), bf16 weights row-major [row][Cin] in global (L2-resident).
// ISA layout: lane l -> M = l&15; half = l>>4; VGPR r: K = (r&4?16:0) + half*8 + (r&3)*2 (+0/1)
static __device__ __forceinline__ bf16x16
load_afrag_g(const bf16_t* W, int Cin, int mbase, int kbase, int lane) {
  const unsigned* row = (const unsigned*)(W + (size_t)(mbase + (lane & 15)) * Cin);
  int kh = (lane >> 4) * 8;
  FragU f;
#pragma unroll
  for (int r = 0; r < 8; ++r) {
    int k = kbase + ((r & 4) ? 16 : 0) + kh + (r & 3) * 2;
    f.u[r] = row[k >> 1];
  }
  return f.v;
}

// B fragment: 32(K) x 16(N), from LDS stored transposed [col][k] (stride S elements, S even).
static __device__ __forceinline__ bf16x16
load_bfrag_lds(const bf16_t* s, int S, int colbase, int kbase, int lane) {
  const unsigned* row = (const unsigned*)(s + (size_t)(colbase + (lane & 15)) * S);
  int kh = kbase + (lane >> 4) * 8;
  FragU f;
#pragma unroll
  for (int r = 0; r < 8; ++r) f.u[r] = row[(kh >> 1) + (r & 3) + ((r & 4) ? 8 : 0)];
  return f.v;
}

// ---- Tensor Data Mover: DMA a 2D fp32 tile (nRows x 128 elems, row stride = strideElems) ----
// D# built per ISA 08_async_tensor 8.3/8.4: group0 {count=1, lds_addr, global_addr, type=2},
// group1 {data_size=4B, tensor_dim0=128, tensor_dim1=nRows, tile 128 x nRows, dim0_stride}.
static __device__ __forceinline__ void
tdm_load_rows_f32(const float* src, int strideElems, int nRows, unsigned ldsByteOff) {
  unsigned long long ga = (unsigned long long)(uintptr_t)src;
  u32x4 g0;
  g0[0] = 1u;                                   // count=1, user descriptor
  g0[1] = ldsByteOff;                           // lds_addr (bytes)
  g0[2] = (unsigned)ga;                         // global_addr[31:0]
  g0[3] = (unsigned)(ga >> 32) | (2u << 30);    // global_addr[56:32] | type=2 ("image")
  i32x8 g1;
  g1[0] = (2 << 16);                            // workgroup_mask=0, data_size=2 (4 bytes)
  g1[1] = (128 << 16);                          // tensor_dim0 = 128 elems (tile-local, no OOB)
  g1[2] = (nRows << 16);                        // tensor_dim1 = nRows
  g1[3] = (128 << 16);                          // tile_dim0 = 128
  g1[4] = nRows;                                // tile_dim1 = nRows, tile_dim2 = 0 (2D)
  g1[5] = strideElems;                          // tensor_dim0_stride[31:0]
  g1[6] = 0;                                    // stride hi / dim1_stride (unused for 2D)
  g1[7] = 0;
  i32x4 z4 = {0, 0, 0, 0};
#if defined(__clang_major__) && (__clang_major__ >= 23)
  i32x8 z8 = {0, 0, 0, 0, 0, 0, 0, 0};
  __builtin_amdgcn_tensor_load_to_lds(g0, g1, z4, z4, z8, 0);
#else
  __builtin_amdgcn_tensor_load_to_lds(g0, g1, z4, z4, 0);
#endif
}

__global__ void f2bf_kernel(const float* __restrict__ s, bf16_t* __restrict__ d, int n) {
  int i = blockIdx.x * 256 + threadIdx.x;
  if (i < n) d[i] = (bf16_t)s[i];
}

// ---------------- KNN: per-batch all points in LDS, per-thread top-8 insertion ----------------
__global__ __launch_bounds__(256) void knn_kernel(const float* __restrict__ pos, int* __restrict__ idx) {
  __shared__ float s_px[2048], s_py[2048], s_pz[2048], s_sq[2048];
  const int b = blockIdx.y;
  const float* P = pos + (size_t)b * 3 * 2048;
  for (int i = threadIdx.x; i < 2048; i += 256) {
    float x = P[i], y = P[2048 + i], z = P[4096 + i];
    s_px[i] = x; s_py[i] = y; s_pz[i] = z;
    s_sq[i] = x * x + y * y + z * z;
  }
  __syncthreads();
  const int n = blockIdx.x * 256 + threadIdx.x;
  float qx = s_px[n], qy = s_py[n], qz = s_pz[n], qs = s_sq[n];
  float bd[8]; int bi[8];
#pragma unroll
  for (int t = 0; t < 8; ++t) { bd[t] = 3.4e38f; bi[t] = 0; }
  for (int m = 0; m < 2048; ++m) {
    float d = qs + s_sq[m] - 2.f * (qx * s_px[m] + qy * s_py[m] + qz * s_pz[m]);
    if (d < bd[7]) {
      int p = 7;
      while (p > 0 && bd[p - 1] > d) { bd[p] = bd[p - 1]; bi[p] = bi[p - 1]; --p; }
      bd[p] = d; bi[p] = m;
    }
  }
  int* o = idx + ((size_t)b * 2048 + n) * 8;
#pragma unroll
  for (int t = 0; t < 8; ++t) o[t] = bi[t];
}

// ---------------- Generic pointwise-conv GEMM: Y[b,c,off] = W(Cout x Cin) * X + epilogue ------
// Block: 256 thr (8 waves). Tile: M=64 (4 mtiles) x N=128 cols (8 ntiles); K chunked by 32.
// X chunk is DMA'd by the TDM into LDS (fp32), then transposed+converted to bf16 for WMMA.
template<int RELU, int HAS_ADD>
__global__ __launch_bounds__(256) void gemm_pw(
    const bf16_t* __restrict__ Wb, const float* __restrict__ bias,
    const float* __restrict__ X1, int C1,
    const float* __restrict__ X2, int C2,
    int Cout, int Ncols,
    const float* __restrict__ addsrc, float alpha,
    float* __restrict__ Y) {
  __shared__ __align__(16) char s_mem[32 * 128 * 4 + 128 * 36 * 2];
  float*  s_stage = (float*)s_mem;                       // [32][128] fp32, LDS offset 0 (TDM dest)
  bf16_t* s_xT    = (bf16_t*)(s_mem + 32 * 128 * 4);     // [128][36] bf16, transposed

  const int Cin = C1 + C2;
  const int tid = threadIdx.x, lane = tid & 31, wid = tid >> 5;
  const int colbase = blockIdx.x * 128;
  const int b = colbase / Ncols;
  const int off0 = colbase - b * Ncols;
  const int mt = wid & 3, ntg = wid >> 2;
  const int mrow0 = blockIdx.y * 64;

  floatx8 acc[4] = {};
  for (int kbase = 0; kbase < Cin; kbase += 32) {
    __syncthreads();
    if (wid == 0) {      // TDM issue + wait are per-wave; one wave drives the DMA
      const float* src;
      if (kbase < C1) src = X1 + ((size_t)b * C1 + kbase) * Ncols + off0;
      else            src = X2 + ((size_t)b * C2 + (kbase - C1)) * Ncols + off0;
      tdm_load_rows_f32(src, Ncols, 32, 0u);
      __builtin_amdgcn_s_wait_tensorcnt(0);
    }
    __syncthreads();
    for (int i = tid; i < 32 * 128; i += 256) {            // transpose + bf16 convert in LDS
      int col = i & 127, kr = i >> 7;
      s_xT[col * 36 + kr] = (bf16_t)s_stage[kr * 128 + col];
    }
    __syncthreads();
    bf16x16 a = load_afrag_g(Wb, Cin, mrow0 + mt * 16, kbase, lane);
#pragma unroll
    for (int j = 0; j < 4; ++j) {
      bf16x16 bb = load_bfrag_lds(s_xT, 36, (ntg * 4 + j) * 16, 0, lane);
      acc[j] = wmma_bf16(a, bb, acc[j]);
    }
  }
#pragma unroll
  for (int j = 0; j < 4; ++j) {
    int col = colbase + (ntg * 4 + j) * 16 + (lane & 15);
    int offc = col - b * Ncols;
#pragma unroll
    for (int r = 0; r < 8; ++r) {
      int c = mrow0 + mt * 16 + (r + 8 * (lane >> 4));
      float v = acc[j][r];
      if (bias) v += bias[c];
      if (RELU) v = fmaxf(v, 0.f);
      size_t oidx = ((size_t)b * Cout + c) * Ncols + offc;
      if (HAS_ADD) v = alpha * (v + addsrc[oidx]);
      Y[oidx] = v;
    }
  }
}

// ---------------- Fused: grouping + pe-MLP + attn-MLP + softmax + aggregation -----------------
// Block handles (b, 8 query points) = 64 (n,k) columns. All D x 64 tiles live in LDS (~108 KB).
__global__ __launch_bounds__(256) void attn_fused(
    const float* __restrict__ pos, const int* __restrict__ idx,
    const float* __restrict__ q, const float* __restrict__ kk, const float* __restrict__ vv,
    const bf16_t* __restrict__ pmw2b, const float* __restrict__ pm_b2,
    const float* __restrict__ pm_w1, const float* __restrict__ pm_b1,
    const float* __restrict__ pm_g, const float* __restrict__ pm_bt,
    const bf16_t* __restrict__ amw1b, const float* __restrict__ am_b1,
    const float* __restrict__ am_g, const float* __restrict__ am_bt,
    const bf16_t* __restrict__ amw2b, const float* __restrict__ am_b2,
    float* __restrict__ agg) {
  extern __shared__ char smem[];
  bf16_t* s_peh = (bf16_t*)smem;                 // [64][72]  pe hidden (bf16)
  bf16_t* s_pe  = s_peh + 64 * 72;               // [64][264] pe        (bf16)
  bf16_t* s_x   = s_pe + 64 * 264;               // [64][264] qk_rel+pe (bf16)
  bf16_t* s_h   = s_x + 64 * 264;                // [64][264] hidden chunk (bf16)
  int* s_idx = (int*)(s_h + 64 * 264);           // [64]

  const int b = blockIdx.y;
  const int n0 = blockIdx.x * 8;
  const int tid = threadIdx.x, lane = tid & 31, wid = tid >> 5;
  const float rs = rsqrtf(1.f + EPSF);

  if (tid < 64) s_idx[tid] = idx[((size_t)b * 2048 + n0 + (tid >> 3)) * 8 + (tid & 7)];
  __syncthreads();

  { // pe hidden layer (K=3, VALU): relu(bn(pm_w1 * pos_rel + pm_b1))
    int col = tid & 63, part = tid >> 6;
    int n = n0 + (col >> 3);
    int j = s_idx[col];
    const float* P = pos + (size_t)b * 3 * 2048;
    float rx = P[n] - P[j], ry = P[2048 + n] - P[2048 + j], rz = P[4096 + n] - P[4096 + j];
#pragma unroll
    for (int h = 0; h < 16; ++h) {
      int hc = part * 16 + h;
      float a = pm_w1[hc * 3] * rx + pm_w1[hc * 3 + 1] * ry + pm_w1[hc * 3 + 2] * rz + pm_b1[hc];
      a = fmaxf(a * (pm_g[hc] * rs) + pm_bt[hc], 0.f);
      s_peh[col * 72 + hc] = (bf16_t)a;
    }
  }
  __syncthreads();

  // pe GEMM: (256x64) * (64x64) via WMMA; 16 mtiles x 4 ntiles over 8 waves
#pragma unroll
  for (int t = 0; t < 8; ++t) {
    int mt = wid * 2 + (t >> 2), nt = t & 3;
    floatx8 acc = {};
#pragma unroll
    for (int ks = 0; ks < 64; ks += 32) {
      bf16x16 a = load_afrag_g(pmw2b, 64, mt * 16, ks, lane);
      bf16x16 bb = load_bfrag_lds(s_peh, 72, nt * 16, ks, lane);
      acc = wmma_bf16(a, bb, acc);
    }
    int col = nt * 16 + (lane & 15);
    int cb = mt * 16 + 8 * (lane >> 4);
    unsigned* dst = (unsigned*)(s_pe + col * 264 + cb);
#pragma unroll
    for (int r = 0; r < 8; r += 2)
      dst[r >> 1] = pack2(acc[r] + pm_b2[cb + r], acc[r + 1] + pm_b2[cb + r + 1]);
  }
  __syncthreads();

  { // build x = q - k_gathered + pe   (q/k gathers hit L2; 192 MB L2 holds them fully)
    int c = tid;
    const float* qb = q + (size_t)b * 256 * 2048;
    const float* kb = kk + (size_t)b * 256 * 2048;
    __builtin_prefetch(vv + (size_t)b * 256 * 2048 + (size_t)c * 2048 + n0, 0, 3); // warm v rows
    float qv[8];
#pragma unroll
    for (int nn = 0; nn < 8; ++nn) qv[nn] = qb[(size_t)c * 2048 + n0 + nn];
    for (int col = 0; col < 64; ++col) {
      int j = s_idx[col];
      float x = qv[col >> 3] - kb[(size_t)c * 2048 + j] + (float)s_pe[col * 264 + c];
      s_x[col * 264 + c] = (bf16_t)x;
    }
  }

  // attn MLP: hidden 512 split into two 256-row chunks; GEMM2 accumulates in registers
  floatx8 acc2[2][4] = {};
  for (int ch = 0; ch < 2; ++ch) {
    __syncthreads();
#pragma unroll
    for (int t = 0; t < 8; ++t) {     // GEMM1 chunk: relu(bn(am_w1 * x + b1)) -> s_h
      int mt = wid * 2 + (t >> 2), nt = t & 3;
      floatx8 acc = {};
#pragma unroll
      for (int ks = 0; ks < 256; ks += 32) {
        bf16x16 a = load_afrag_g(amw1b, 256, ch * 256 + mt * 16, ks, lane);
        bf16x16 bb = load_bfrag_lds(s_x, 264, nt * 16, ks, lane);
        acc = wmma_bf16(a, bb, acc);
      }
      int col = nt * 16 + (lane & 15);
      int cb = mt * 16 + 8 * (lane >> 4);
      unsigned* dst = (unsigned*)(s_h + col * 264 + cb);
#pragma unroll
      for (int r = 0; r < 8; r += 2) {
        int h0 = ch * 256 + cb + r, h1 = h0 + 1;
        float u0 = fmaxf((acc[r]     + am_b1[h0]) * (am_g[h0] * rs) + am_bt[h0], 0.f);
        float u1 = fmaxf((acc[r + 1] + am_b1[h1]) * (am_g[h1] * rs) + am_bt[h1], 0.f);
        dst[r >> 1] = pack2(u0, u1);
      }
    }
    __syncthreads();
#pragma unroll
    for (int mt2 = 0; mt2 < 2; ++mt2)  // GEMM2 accumulate: am_w2 * h
#pragma unroll
      for (int nt = 0; nt < 4; ++nt)
#pragma unroll
        for (int ks = 0; ks < 256; ks += 32) {
          bf16x16 a = load_afrag_g(amw2b, 512, (wid * 2 + mt2) * 16, ch * 256 + ks, lane);
          bf16x16 bb = load_bfrag_lds(s_h, 264, nt * 16, ks, lane);
          acc2[mt2][nt] = wmma_bf16(a, bb, acc2[mt2][nt]);
        }
  }

  // softmax over the 8-neighbor groups (shfl butterflies across 8 lanes) + aggregation
  const float* vb = vv + (size_t)b * 256 * 2048;
#pragma unroll
  for (int mt2 = 0; mt2 < 2; ++mt2) {
    int cb = (wid * 2 + mt2) * 16 + 8 * (lane >> 4);
#pragma unroll
    for (int nt = 0; nt < 4; ++nt) {
      int col = nt * 16 + (lane & 15);
      int nq = n0 + (col >> 3);
      int j = s_idx[col];
      float outv[8];
#pragma unroll
      for (int r = 0; r < 8; ++r) {
        int c = cb + r;
        float w = acc2[mt2][nt][r] + am_b2[c];
        float mx = w;
        mx = fmaxf(mx, __shfl_xor(mx, 1, 32));
        mx = fmaxf(mx, __shfl_xor(mx, 2, 32));
        mx = fmaxf(mx, __shfl_xor(mx, 4, 32));
        float e = __expf(w - mx);
        float s = e;
        s += __shfl_xor(s, 1, 32);
        s += __shfl_xor(s, 2, 32);
        s += __shfl_xor(s, 4, 32);
        float p = e / s;
        float vg = vb[(size_t)c * 2048 + j] + (float)s_pe[col * 264 + c];
        float contrib = p * vg;
        contrib += __shfl_xor(contrib, 1, 32);
        contrib += __shfl_xor(contrib, 2, 32);
        contrib += __shfl_xor(contrib, 4, 32);
        outv[r] = contrib;
      }
      if ((lane & 7) == 0) {
#pragma unroll
        for (int r = 0; r < 8; ++r)
          agg[((size_t)b * 256 + cb + r) * 2048 + nq] = outv[r];
      }
    }
  }
}

extern "C" void kernel_launch(void* const* d_in, const int* in_sizes, int n_in,
                              void* d_out, int out_size, void* d_ws, size_t ws_size,
                              hipStream_t stream) {
  (void)in_sizes; (void)n_in; (void)out_size; (void)ws_size;
  const float* pos   = (const float*)d_in[0];
  const float* fea   = (const float*)d_in[2];
  const float* prev  = (const float*)d_in[3];
  const float* wq    = (const float*)d_in[4];
  const float* wk    = (const float*)d_in[5];
  const float* wv    = (const float*)d_in[6];
  const float* bv    = (const float*)d_in[7];
  const float* vm_w1 = (const float*)d_in[8];
  const float* vm_b1 = (const float*)d_in[9];
  const float* vm_w2 = (const float*)d_in[10];
  const float* vm_b2 = (const float*)d_in[11];
  const float* pm_w1 = (const float*)d_in[12];
  const float* pm_b1 = (const float*)d_in[13];
  const float* pm_g  = (const float*)d_in[14];
  const float* pm_bt = (const float*)d_in[15];
  const float* pm_w2 = (const float*)d_in[16];
  const float* pm_b2 = (const float*)d_in[17];
  const float* am_w1 = (const float*)d_in[18];
  const float* am_b1 = (const float*)d_in[19];
  const float* am_g  = (const float*)d_in[20];
  const float* am_bt = (const float*)d_in[21];
  const float* am_w2 = (const float*)d_in[22];
  const float* am_b2 = (const float*)d_in[23];
  const float* le_w  = (const float*)d_in[24];
  const float* le_b  = (const float*)d_in[25];

  char* ws = (char*)d_ws;
  size_t off = 0;
  auto take = [&](size_t bytes) { char* p = ws + off; off = (off + bytes + 255) & ~(size_t)255; return p; };
  int*   idxb = (int*)  take(8ull * 2048 * 8 * 4);
  float* hid1 = (float*)take(8ull * 128 * 2048 * 4);
  float* v0   = (float*)take(8ull * 128 * 2048 * 4);
  float* qb   = (float*)take(8ull * 256 * 2048 * 4);
  float* kb   = (float*)take(8ull * 256 * 2048 * 4);
  float* vvb  = (float*)take(8ull * 256 * 2048 * 4);
  float* aggb = (float*)take(8ull * 256 * 2048 * 4);
  bf16_t* vmw1b = (bf16_t*)take(128 * 256 * 2);
  bf16_t* vmw2b = (bf16_t*)take(128 * 128 * 2);
  bf16_t* wqb   = (bf16_t*)take(256 * 128 * 2);
  bf16_t* wkb   = (bf16_t*)take(256 * 128 * 2);
  bf16_t* wvb   = (bf16_t*)take(256 * 128 * 2);
  bf16_t* pmw2b = (bf16_t*)take(256 * 64 * 2);
  bf16_t* amw1b = (bf16_t*)take(512 * 256 * 2);
  bf16_t* amw2b = (bf16_t*)take(256 * 512 * 2);
  bf16_t* lewb  = (bf16_t*)take(128 * 256 * 2);

  auto conv = [&](const float* s, bf16_t* d, int n) {
    f2bf_kernel<<<dim3((n + 255) / 256), dim3(256), 0, stream>>>(s, d, n);
  };
  conv(vm_w1, vmw1b, 128 * 256);
  conv(vm_w2, vmw2b, 128 * 128);
  conv(wq,    wqb,   256 * 128);
  conv(wk,    wkb,   256 * 128);
  conv(wv,    wvb,   256 * 128);
  conv(pm_w2, pmw2b, 256 * 64);
  conv(am_w1, amw1b, 512 * 256);
  conv(am_w2, amw2b, 256 * 512);
  conv(le_w,  lewb,  128 * 256);

  knn_kernel<<<dim3(8, 8), 256, 0, stream>>>(pos, idxb);

  // value_mlp: relu(vm_w1*[fea;prev]+b1) -> vm_w2*h+b2 = v0 (identity)
  gemm_pw<1, 0><<<dim3(128, 2), 256, 0, stream>>>(vmw1b, vm_b1, fea, 128, prev, 128, 128, 2048, nullptr, 0.f, hid1);
  gemm_pw<0, 0><<<dim3(128, 2), 256, 0, stream>>>(vmw2b, vm_b2, hid1, 128, nullptr, 0, 128, 2048, nullptr, 0.f, v0);
  // q/k/v projections
  gemm_pw<0, 0><<<dim3(128, 4), 256, 0, stream>>>(wqb, nullptr, fea, 128, nullptr, 0, 256, 2048, nullptr, 0.f, qb);
  gemm_pw<0, 0><<<dim3(128, 4), 256, 0, stream>>>(wkb, nullptr, prev, 128, nullptr, 0, 256, 2048, nullptr, 0.f, kb);
  gemm_pw<0, 0><<<dim3(128, 4), 256, 0, stream>>>(wvb, bv, v0, 128, nullptr, 0, 256, 2048, nullptr, 0.f, vvb);

  size_t smem = (size_t)(64 * 72 + 3 * 64 * 264) * 2 + 256;
  (void)hipFuncSetAttribute(reinterpret_cast<const void*>(attn_fused),
                            hipFuncAttributeMaxDynamicSharedMemorySize, (int)smem);
  attn_fused<<<dim3(256, 8), 256, smem, stream>>>(pos, idxb, qb, kb, vvb,
      pmw2b, pm_b2, pm_w1, pm_b1, pm_g, pm_bt,
      amw1b, am_b1, am_g, am_bt, amw2b, am_b2, aggb);

  // final: 0.5 * (le_w*agg + le_b + identity)
  gemm_pw<0, 1><<<dim3(128, 2), 256, 0, stream>>>(lewb, le_b, aggb, 256, nullptr, 0, 128, 2048, v0, 0.5f, (float*)d_out);
}
